// GCN_Net_22385369546944
// MI455X (gfx1250) — matrix operand order repaired and checked by reading, not compiled, not run on hardware
//
#include <hip/hip_runtime.h>

#define IN_CH 500
#define HID   64
#define OUTC  40
#define KPAD1 512              // IN_CH padded; k in [500,512) zero in Wt1

typedef __attribute__((ext_vector_type(16))) _Float16 v16h;
typedef __attribute__((ext_vector_type(8)))  _Float16 v8h;
typedef __attribute__((ext_vector_type(8)))  float    v8f;
typedef __attribute__((ext_vector_type(4)))  float    f4;

union U16H { v16h v; v8h h[2]; };

// 32-byte B fragment: two global_load_b128 (16B-aligned addresses).
__device__ __forceinline__ v16h load_b16(const _Float16* p) {
    U16H u;
    u.h[0] = *(const v8h*)(p);
    u.h[1] = *(const v8h*)(p + 16);
    return u.v;
}

// Per-lane 16x32 f16 A fragment from an f32 row (optionally ReLU'd).
// ISA layout (CDNA5 ISA 7.12.2): lanes 0-15 own K={k0..k0+7, k0+16..k0+23},
// lanes 16-31 own K={k0+8..k0+15, k0+24..k0+31}. Aligned float4 quads.
// GUARD: zero-fill quads whose start >= kvalid (only needed for the K tail;
// elsewhere the zero-padded B columns make any extra A data harmless).
template <bool RELU, bool GUARD>
__device__ __forceinline__ v16h load_a_f32(const float* rowp, int k0, int kvalid) {
    const int lane = threadIdx.x & 31;
    const int half = lane >> 4;
    const int s0 = k0 + half * 8;
    const int s1 = k0 + 16 + half * 8;
    const int starts[4] = { s0, s0 + 4, s1, s1 + 4 };
    v16h a;
#pragma unroll
    for (int g = 0; g < 4; ++g) {
        f4 q = {0.f, 0.f, 0.f, 0.f};
        if (!GUARD || starts[g] < kvalid)
            q = *(const f4*)(rowp + starts[g]);
        if (RELU) {
            q.x = fmaxf(q.x, 0.f); q.y = fmaxf(q.y, 0.f);
            q.z = fmaxf(q.z, 0.f); q.w = fmaxf(q.w, 0.f);
        }
        a[4 * g + 0] = (_Float16)q.x;
        a[4 * g + 1] = (_Float16)q.y;
        a[4 * g + 2] = (_Float16)q.z;
        a[4 * g + 3] = (_Float16)q.w;
    }
    return a;
}

#define WMMA(A, B, C) \
    __builtin_amdgcn_wmma_f32_16x16x32_f16(false, (A), false, (B), (short)0, (C), false, false)

// ---- One-time weight prep: f32 [K][N] -> f16 transposed [N][Kpad], zero pad
__global__ void k_prepW1(const float* __restrict__ W, _Float16* __restrict__ Wt) {
    int idx = blockIdx.x * blockDim.x + threadIdx.x;   // 64 * 512
    if (idx < HID * KPAD1) {
        int nn = idx >> 9, k = idx & (KPAD1 - 1);
        Wt[idx] = (_Float16)((k < IN_CH) ? W[k * HID + nn] : 0.f);
    }
}
__global__ void k_prepW2(const float* __restrict__ W, _Float16* __restrict__ Wt) {
    int idx = blockIdx.x * blockDim.x + threadIdx.x;   // 48 * 64
    if (idx < 48 * HID) {
        int nn = idx >> 6, k = idx & 63;
        Wt[idx] = (_Float16)((nn < OUTC) ? W[k * OUTC + nn] : 0.f);
    }
}

// ---------------- GEMM1: xw[n,64] = x[n,500] @ W[500,64] -------------------
// 8 waves/block; wave w computes rows [blk*128 + 16w, +16). B fragments come
// straight from the L2-resident f16 Wt1[64][512]; no LDS, no syncthreads.
__global__ void __launch_bounds__(256) k_gemm1(const float* __restrict__ X,
                                               const _Float16* __restrict__ Wt,
                                               float* __restrict__ XW, int n) {
    const int tid  = threadIdx.x;
    const int wave = tid >> 5;
    const int lane = tid & 31;
    const int half = lane >> 4;
    const int r    = lane & 15;
    const int rowBase = blockIdx.x * 128 + wave * 16;
    const int row  = min(rowBase + r, n - 1);        // clamp: loads always valid
    const float* rowp = X + (long)row * IN_CH;
    const _Float16* bbase = Wt + r * KPAD1 + half * 8;

    v8f acc0 = {}, acc1 = {}, acc2 = {}, acc3 = {};
#pragma unroll
    for (int k0 = 0; k0 < 480; k0 += 32) {           // steady state: branchless
        v16h a = load_a_f32<false, false>(rowp, k0, IN_CH);
        const _Float16* bp = bbase + k0;
        v16h b0 = load_b16(bp);
        v16h b1 = load_b16(bp + 16 * KPAD1);
        v16h b2 = load_b16(bp + 32 * KPAD1);
        v16h b3 = load_b16(bp + 48 * KPAD1);
        acc0 = WMMA(a, b0, acc0);
        acc1 = WMMA(a, b1, acc1);
        acc2 = WMMA(a, b2, acc2);
        acc3 = WMMA(a, b3, acc3);
    }
    {                                                // K tail (k0=480): guard reads
        const int k0 = 480;
        v16h a = load_a_f32<false, true>(rowp, k0, IN_CH);
        const _Float16* bp = bbase + k0;
        v16h b0 = load_b16(bp);
        v16h b1 = load_b16(bp + 16 * KPAD1);
        v16h b2 = load_b16(bp + 32 * KPAD1);
        v16h b3 = load_b16(bp + 48 * KPAD1);
        acc0 = WMMA(a, b0, acc0);
        acc1 = WMMA(a, b1, acc1);
        acc2 = WMMA(a, b2, acc2);
        acc3 = WMMA(a, b3, acc3);
    }
#pragma unroll
    for (int v = 0; v < 8; ++v) {
        int orow = rowBase + v + half * 8;           // D layout: VGPR v -> M=v+8*half
        if (orow < n) {
            float* op = XW + (long)orow * HID;
            op[     r] = acc0[v];
            op[16 + r] = acc1[v];
            op[32 + r] = acc2[v];
            op[48 + r] = acc3[v];
        }
    }
}

// ---------------- GEMM2: hw[n,40] = relu(h[n,64]) @ W[64,40] ---------------
__global__ void __launch_bounds__(256) k_gemm2(const float* __restrict__ H,
                                               const _Float16* __restrict__ Wt,
                                               float* __restrict__ HW, int n) {
    const int tid  = threadIdx.x;
    const int wave = tid >> 5;
    const int lane = tid & 31;
    const int half = lane >> 4;
    const int r    = lane & 15;
    const int rowBase = blockIdx.x * 128 + wave * 16;
    const int row  = min(rowBase + r, n - 1);
    const float* rowp = H + (long)row * HID;
    const _Float16* bbase = Wt + r * HID + half * 8;

    v8f acc0 = {}, acc1 = {}, acc2 = {};
#pragma unroll
    for (int k0 = 0; k0 < HID; k0 += 32) {
        v16h a = load_a_f32<true, false>(rowp, k0, HID);   // fused ReLU
        const _Float16* bp = bbase + k0;
        v16h b0 = load_b16(bp);
        v16h b1 = load_b16(bp + 16 * HID);
        v16h b2 = load_b16(bp + 32 * HID);
        acc0 = WMMA(a, b0, acc0);
        acc1 = WMMA(a, b1, acc1);
        acc2 = WMMA(a, b2, acc2);
    }
#pragma unroll
    for (int v = 0; v < 8; ++v) {
        int orow = rowBase + v + half * 8;
        if (orow < n) {
            float* op = HW + (long)orow * OUTC;
            op[     r] = acc0[v];
            op[16 + r] = acc1[v];
            if (32 + r < OUTC) op[32 + r] = acc2[v];
        }
    }
}

// ---------------- Degree / normalization ----------------------------------
__global__ void k_fill1(float* __restrict__ d, int n) {
    int i = blockIdx.x * blockDim.x + threadIdx.x;
    if (i < n) d[i] = 1.0f;                          // self-loop contribution
}
__global__ void k_deg(const int* __restrict__ ei, int E, float* __restrict__ deg) {
    int e = blockIdx.x * blockDim.x + threadIdx.x;
    if (e < E) atomicAdd(deg + ei[E + e], 1.0f);     // dst row of edge_index
}
__global__ void k_dinv(float* __restrict__ d, int n) {
    int i = blockIdx.x * blockDim.x + threadIdx.x;
    if (i < n) d[i] = rsqrtf(fmaxf(d[i], 1.0f));
}

// ---------------- Bias + self-loop init, edge scatter-adds ----------------
__global__ void k_init_h(float* __restrict__ h, const float* __restrict__ xw,
                         const float* __restrict__ dinv, const float* __restrict__ b,
                         int n) {
    unsigned g = (unsigned)blockIdx.x * blockDim.x + threadIdx.x;
    if (g < (unsigned)n * HID) {
        unsigned i = g >> 6, c = g & 63;
        float di = dinv[i];
        h[g] = b[c] + di * di * xw[g];
    }
}
__global__ void k_agg1(const int* __restrict__ ei, int E,
                       const float* __restrict__ dinv, const float* __restrict__ xw,
                       float* __restrict__ h) {
    unsigned g = (unsigned)blockIdx.x * blockDim.x + threadIdx.x;
    if (g < (unsigned)E * HID) {
        unsigned e = g >> 6, c = g & 63;          // 2 waves/edge, uniform e per wave
        int s = ei[e], d = ei[E + e];
        float w = dinv[s] * dinv[d];
        atomicAdd(h + ((unsigned)d << 6) + c, w * xw[((unsigned)s << 6) + c]);
    }
}
__global__ void k_init_out(float* __restrict__ out, const float* __restrict__ hw,
                           const float* __restrict__ dinv, const float* __restrict__ b,
                           int n) {
    unsigned g = (unsigned)blockIdx.x * blockDim.x + threadIdx.x;
    if (g < (unsigned)n * OUTC) {
        unsigned i = g / OUTC, c = g - i * OUTC;
        float di = dinv[i];
        out[g] = b[c] + di * di * hw[g];
    }
}
__global__ void k_agg2(const int* __restrict__ ei, int E,
                       const float* __restrict__ dinv, const float* __restrict__ hw,
                       float* __restrict__ out) {
    unsigned g = (unsigned)blockIdx.x * blockDim.x + threadIdx.x;
    if (g < (unsigned)E * OUTC) {
        unsigned e = g / OUTC, c = g - e * OUTC;
        int s = ei[e], d = ei[E + e];
        float w = dinv[s] * dinv[d];
        atomicAdd(out + (unsigned)d * OUTC + c, w * hw[(unsigned)s * OUTC + c]);
    }
}

extern "C" void kernel_launch(void* const* d_in, const int* in_sizes, int n_in,
                              void* d_out, int out_size, void* d_ws, size_t ws_size,
                              hipStream_t stream) {
    (void)n_in; (void)out_size;
    const float* x  = (const float*)d_in[0];
    const int*   ei = (const int*)  d_in[1];
    const float* Wc = (const float*)d_in[2];
    const float* bc = (const float*)d_in[3];
    const float* Wk = (const float*)d_in[4];
    const float* bk = (const float*)d_in[5];
    float* out = (float*)d_out;

    const int n = in_sizes[0] / IN_CH;
    const int E = in_sizes[1] / 2;

    // Workspace: [dinv: n][xw: n*64 (reused as hw)][h_pre: n*64][Wt1][Wt2]
    float* dinv = (float*)d_ws;
    float* xw   = dinv + n;
    float* hpre = xw + (size_t)n * HID;
    size_t fbytes = ((size_t)n * (1 + 2 * HID) * sizeof(float) + 255) & ~(size_t)255;
    _Float16* Wt1 = (_Float16*)((char*)d_ws + fbytes);
    _Float16* Wt2 = Wt1 + HID * KPAD1;
    size_t need = fbytes + (HID * KPAD1 + 48 * HID) * sizeof(_Float16);
    if (ws_size < need) return;

    const int B = 256;
    auto cdiv = [](long a, long b) { return (unsigned)((a + b - 1) / b); };

    k_prepW1  <<<cdiv(HID * KPAD1, B),    B, 0, stream>>>(Wc, Wt1);
    k_prepW2  <<<cdiv(48 * HID, B),       B, 0, stream>>>(Wk, Wt2);
    k_fill1   <<<cdiv(n, B),              B, 0, stream>>>(dinv, n);
    k_deg     <<<cdiv(E, B),              B, 0, stream>>>(ei, E, dinv);
    k_dinv    <<<cdiv(n, B),              B, 0, stream>>>(dinv, n);
    k_gemm1   <<<cdiv(n, 128),          256, 0, stream>>>(x, Wt1, xw, n);
    k_init_h  <<<cdiv((long)n * HID, B),  B, 0, stream>>>(hpre, xw, dinv, bc, n);
    k_agg1    <<<cdiv((long)E * HID, B),  B, 0, stream>>>(ei, E, dinv, xw, hpre);
    k_gemm2   <<<cdiv(n, 128),          256, 0, stream>>>(hpre, Wt2, xw, n);   // xw->hw
    k_init_out<<<cdiv((long)n * OUTC, B), B, 0, stream>>>(out, xw, dinv, bk, n);
    k_agg2    <<<cdiv((long)E * OUTC, B), B, 0, stream>>>(ei, E, dinv, xw, out);
}
